// RGCN_42013370089999
// MI455X (gfx1250) — compile-verified, matrix operand order checked by reading, not run on hardware
//
#include <hip/hip_runtime.h>
#include <hip/hip_bf16.h>

// ---------------------------------------------------------------------------
// RGCN (2-layer, basis decomposition) for MI455X / gfx1250.
// Scatter-mean via f32 global atomics, then one fused GEMM per layer:
//   out = [agg/cnt | h] (N x 3200) @ [W; root] (3200 x 128) + bias
// on v_wmma_f32_16x16x32_bf16 (fp32 accum). B pre-packed lane-major in the
// exact WMMA fragment layout (2x global_load_b128 per fragment, all 8
// fragments preloaded per k-step with staggered s_wait_loadcnt); A staged
// through double-buffered LDS in fragment-contiguous order (2x ds_load_b128).
// f32->bf16 uses truncation packing (1 VALU op/pair via v_perm/v_and_or).
// ---------------------------------------------------------------------------

#define DDIM   128
#define RREL   24
#define BBAS   8
#define KT_AGG 96            // K-tiles (of 32) sourced from agg
#define KTILES 100           // 3200 / 32
#define ASTRIDE 20           // LDS row stride in dwords (16B-aligned, conflict-free)

typedef __attribute__((ext_vector_type(16))) __bf16        v16bf;
typedef __attribute__((ext_vector_type(8)))  float         v8f;
typedef __attribute__((ext_vector_type(4)))  unsigned int  v4u;

union ABFrag { unsigned int u[8]; v4u q[2]; v16bf v; };

// Truncating f32->bf16 pair pack: single v_perm_b32 / v_and_or_b32.
__device__ __forceinline__ unsigned int pack2(float lo, float hi) {
    unsigned int ulo = __builtin_bit_cast(unsigned int, lo);
    unsigned int uhi = __builtin_bit_cast(unsigned int, hi);
    return (uhi & 0xFFFF0000u) | (ulo >> 16);
}
// pair-index permutation so each lane's 8 fragment dwords are contiguous
__device__ __forceinline__ int ppos(int p) {           // swap bits 2 <-> 3
    return (p & 3) | ((p & 4) << 1) | ((p & 8) >> 1);
}

// ---------------------------------------------------------------- embedding
__global__ __launch_bounds__(256)
void rgcn_gather(float* __restrict__ h, const float* __restrict__ emb,
                 const int* __restrict__ x, int n) {
    int t = blockIdx.x * 256 + threadIdx.x;           // n*32 threads, float4 each
    int row = t >> 5, lane = t & 31;
    if (row >= n) return;
    const float4* s = (const float4*)(emb + (size_t)x[row] * DDIM);
    float4* d = (float4*)(h + (size_t)row * DDIM);
    d[lane] = s[lane];
}

// ----------------------------------------------------- edge scatter (atomic)
__global__ __launch_bounds__(256)
void rgcn_edge_agg(float* __restrict__ agg, float* __restrict__ cnt,
                   const float* __restrict__ h, const int* __restrict__ ei,
                   const int* __restrict__ et, int e) {
    int t = blockIdx.x * 256 + threadIdx.x;           // one wave per edge
    int edge = t >> 5, lane = t & 31;
    if (edge >= e) return;
    int src = ei[edge];
    int dst = ei[e + edge];
    int rel = et[edge];
    size_t seg = (size_t)dst * RREL + rel;
    const float4* hs = (const float4*)(h + (size_t)src * DDIM);
    float4 v = hs[lane];
    float* a = agg + seg * DDIM + lane * 4;
    atomicAdd(a + 0, v.x);
    atomicAdd(a + 1, v.y);
    atomicAdd(a + 2, v.z);
    atomicAdd(a + 3, v.w);
    if (lane == 0) atomicAdd(cnt + seg, 1.0f);
}

// --------------------------------------------------- counts -> 1/max(cnt,1)
__global__ __launch_bounds__(256)
void rgcn_invcnt(float* __restrict__ cnt, int nr) {
    int i = blockIdx.x * 256 + threadIdx.x;
    if (i >= nr) return;
    cnt[i] = 1.0f / fmaxf(cnt[i], 1.0f);
}

// ------------------- pack Wcat = [comp@bases ; root], lane-major fragments:
// Wb[((kt*8 + nt)*32 + lane)*8 + v] = pack(bf16 W[k0][col], bf16 W[k0+1][col])
// k0 = kt*32 + (v<4?0:16) + 8*(lane>=16) + 2*(v&3), col = nt*16 + (lane&15)
__device__ __forceinline__ float wcat_val(int k, int col,
                                          const float* __restrict__ comp,
                                          const float* __restrict__ bases,
                                          const float* __restrict__ root) {
    if (k < RREL * DDIM) {
        int r = k >> 7, i = k & 127;
        float s = 0.f;
#pragma unroll
        for (int b = 0; b < BBAS; ++b)
            s += comp[r * BBAS + b] * bases[(size_t)b * DDIM * DDIM + i * DDIM + col];
        return s;
    }
    return root[(size_t)(k - RREL * DDIM) * DDIM + col];
}

__global__ __launch_bounds__(256)
void rgcn_wprep(unsigned int* __restrict__ Wb, const float* __restrict__ comp,
                const float* __restrict__ bases, const float* __restrict__ root) {
    int t  = blockIdx.x * 256 + threadIdx.x;          // KTILES*8*8*32 threads
    int ln = t & 31;
    int v  = (t >> 5) & 7;
    int nt = (t >> 8) & 7;
    int kt = t >> 11;
    if (kt >= KTILES) return;
    int hh  = ln >> 4, nn = ln & 15;
    int col = nt * 16 + nn;
    int k0  = kt * 32 + ((v & 4) ? 16 : 0) + hh * 8 + (v & 3) * 2;
    float w0 = wcat_val(k0,     col, comp, bases, root);
    float w1 = wcat_val(k0 + 1, col, comp, bases, root);
    Wb[((size_t)(kt * 8 + nt) * 32 + ln) * 8 + v] = pack2(w0, w1);
}

// --------------------------------------------------------------- WMMA GEMM
// 256 threads = 8 waves; 128x128 output tile per block; wave w -> 16 rows,
// all 8 n-tiles. K loop: 100 steps of 32, double-buffered LDS A staging with
// fused mean division and fp32 -> bf16 conversion. All 8 B fragments of a
// k-step are preloaded so the backend staggers s_wait_loadcnt across WMMAs.
__global__ __launch_bounds__(256)
void rgcn_gemm(float* __restrict__ out, const float* __restrict__ agg,
               const float* __restrict__ invc, const float* __restrict__ hin,
               const unsigned int* __restrict__ Wb, const float* __restrict__ bias,
               int n, int relu) {
    __shared__ unsigned int ldsA[2][128 * ASTRIDE];   // 2 x 10240 B

    const int tid     = threadIdx.x;
    const int w       = tid >> 5;                     // wave 0..7
    const int ln      = tid & 31;                     // lane 0..31
    const int mm      = ln & 15;
    const int hh      = ln >> 4;
    const int rowBase = blockIdx.x * 128;

    const int lr   = tid >> 1;                        // staging row 0..127
    const int half = tid & 1;                         // 16-float k-half

    // ---- A tile staging: 128x32 fp32 -> bf16, fragment-contiguous order ---
    auto stage = [&](int kt, int buf) {
        int row = rowBase + lr;
        float4 q0 = {0,0,0,0}, q1 = q0, q2 = q0, q3 = q0;
        float scale = 0.f;
        if (row < n) {
            const float4* s4;
            if (kt < KT_AGG) {
                scale = invc[(size_t)row * RREL + (kt >> 2)];
                s4 = (const float4*)(agg + (size_t)row * (RREL * DDIM)
                                     + kt * 32 + half * 16);
            } else {
                scale = 1.0f;
                s4 = (const float4*)(hin + (size_t)row * DDIM
                                     + (kt - KT_AGG) * 32 + half * 16);
            }
            q0 = s4[0]; q1 = s4[1]; q2 = s4[2]; q3 = s4[3];
        }
        unsigned int* d = &ldsA[buf][lr * ASTRIDE];
        const int p0 = half * 8;                      // pair indices p0..p0+7
        d[ppos(p0 + 0)] = pack2(q0.x * scale, q0.y * scale);
        d[ppos(p0 + 1)] = pack2(q0.z * scale, q0.w * scale);
        d[ppos(p0 + 2)] = pack2(q1.x * scale, q1.y * scale);
        d[ppos(p0 + 3)] = pack2(q1.z * scale, q1.w * scale);
        d[ppos(p0 + 4)] = pack2(q2.x * scale, q2.y * scale);
        d[ppos(p0 + 5)] = pack2(q2.z * scale, q2.w * scale);
        d[ppos(p0 + 6)] = pack2(q3.x * scale, q3.y * scale);
        d[ppos(p0 + 7)] = pack2(q3.z * scale, q3.w * scale);
    };

    v8f acc[8] = {};                                  // 8 n-tiles of 16x16 f32

    stage(0, 0);
    __syncthreads();

    for (int kt = 0; kt < KTILES; ++kt) {
        const int cur = kt & 1;

        // ---- A fragment: two ds_load_b128 (contiguous, bank-conflict-free)
        ABFrag a;
        {
            const v4u* ap = (const v4u*)&ldsA[cur][(w * 16 + mm) * ASTRIDE + hh * 8];
            a.q[0] = ap[0];
            a.q[1] = ap[1];
        }

        // ---- preload ALL 8 B fragments (one big clause, staggered waits) --
        ABFrag b[8];
        {
            const v4u* bp = (const v4u*)(Wb + ((size_t)(kt * 8) * 32 + ln) * 8);
#pragma unroll
            for (int nt = 0; nt < 8; ++nt) {
                b[nt].q[0] = bp[nt * 64 + 0];         // 32 lanes * 2 v4u = 64
                b[nt].q[1] = bp[nt * 64 + 1];
            }
        }

        // prefetch next kt's B panel (emits global_prefetch_b8)
        if (kt + 1 < KTILES)
            __builtin_prefetch(Wb + (size_t)(kt + 1) * 8 * 8 * 32 + tid, 0, 1);

        // ---- stage next tile while B loads are in flight ------------------
        if (kt + 1 < KTILES) stage(kt + 1, cur ^ 1);

        // ---- 8 WMMAs ------------------------------------------------------
#pragma unroll
        for (int nt = 0; nt < 8; ++nt) {
            acc[nt] = __builtin_amdgcn_wmma_f32_16x16x32_bf16(
                false, a.v, false, b[nt].v, (short)0, acc[nt], false, false);
        }
        __syncthreads();                              // one barrier / iteration
    }

    // ---- epilogue: bias (+ReLU), ISA 32-bit C/D layout --------------------
#pragma unroll
    for (int nt = 0; nt < 8; ++nt) {
        int   col = nt * 16 + mm;
        float bc  = bias[col];
#pragma unroll
        for (int j = 0; j < 8; ++j) {
            int row = rowBase + w * 16 + j + 8 * hh;
            if (row < n) {
                float vf = acc[nt][j] + bc;
                if (relu) vf = fmaxf(vf, 0.0f);
                out[(size_t)row * DDIM + col] = vf;
            }
        }
    }
}

// ---------------------------------------------------------------------------
extern "C" void kernel_launch(void* const* d_in, const int* in_sizes, int n_in,
                              void* d_out, int out_size, void* d_ws, size_t ws_size,
                              hipStream_t stream) {
    const int*   x     = (const int*)d_in[0];
    const int*   ei    = (const int*)d_in[1];       // [2, E] flat
    const int*   et    = (const int*)d_in[2];
    const float* emb   = (const float*)d_in[3];
    const float* comp1 = (const float*)d_in[4];
    const float* bas1  = (const float*)d_in[5];
    const float* root1 = (const float*)d_in[6];
    const float* bias1 = (const float*)d_in[7];
    const float* comp2 = (const float*)d_in[8];
    const float* bas2  = (const float*)d_in[9];
    const float* root2 = (const float*)d_in[10];
    const float* bias2 = (const float*)d_in[11];

    const int n = in_sizes[0];                      // 50000 nodes
    const int e = in_sizes[2];                      // 1.6M edges
    float* out = (float*)d_out;

    // ---- workspace carve-out ----------------------------------------------
    char* ws = (char*)d_ws;
    size_t off = 0;
    auto take = [&](size_t bytes) {
        void* p = ws + off;
        off += (bytes + 255) & ~(size_t)255;
        return p;
    };
    float*        h   = (float*)take((size_t)n * DDIM * 4);
    float*        h2  = (float*)take((size_t)n * DDIM * 4);
    float*        agg = (float*)take((size_t)n * RREL * DDIM * 4);
    float*        cnt = (float*)take((size_t)n * RREL * 4);          // becomes 1/cnt
    unsigned int* Wb  = (unsigned int*)take((size_t)KTILES * 8 * 8 * 32 * 4);
    (void)ws_size; (void)n_in; (void)out_size;

    const int nr      = n * RREL;
    const int gGather = (n * 32 + 255) / 256;
    const int gEdge   = (e * 32 + 255) / 256;
    const int gInv    = (nr + 255) / 256;
    const int gWprep  = (KTILES * 8 * 8 * 32 + 255) / 256;
    const int gGemm   = (n + 127) / 128;

    // ---- layer 1 ----------------------------------------------------------
    rgcn_gather<<<gGather, 256, 0, stream>>>(h, emb, x, n);
    hipMemsetAsync(agg, 0, (size_t)n * RREL * DDIM * 4, stream);
    hipMemsetAsync(cnt, 0, (size_t)nr * 4, stream);
    rgcn_edge_agg<<<gEdge, 256, 0, stream>>>(agg, cnt, h, ei, et, e);
    rgcn_invcnt<<<gInv, 256, 0, stream>>>(cnt, nr);
    rgcn_wprep<<<gWprep, 256, 0, stream>>>(Wb, comp1, bas1, root1);
    rgcn_gemm<<<gGemm, 256, 0, stream>>>(h2, agg, cnt, h, Wb, bias1, n, /*relu=*/1);

    // ---- layer 2 ----------------------------------------------------------
    hipMemsetAsync(agg, 0, (size_t)n * RREL * DDIM * 4, stream);
    hipMemsetAsync(cnt, 0, (size_t)nr * 4, stream);
    rgcn_edge_agg<<<gEdge, 256, 0, stream>>>(agg, cnt, h2, ei, et, e);
    rgcn_invcnt<<<gInv, 256, 0, stream>>>(cnt, nr);
    rgcn_wprep<<<gWprep, 256, 0, stream>>>(Wb, comp2, bas2, root2);
    rgcn_gemm<<<gGemm, 256, 0, stream>>>(out, agg, cnt, h2, Wb, bias2, n, /*relu=*/0);
}
